// GAT_18751827214390
// MI455X (gfx1250) — matrix operand order repaired and checked by reading, not compiled, not run on hardware
//
#include <hip/hip_runtime.h>

// ---------------------------------------------------------------------------
// Types for CDNA5 WMMA (wave32, 16x16x32 bf16 -> f32)
// ---------------------------------------------------------------------------
typedef __bf16 bf16_t;
typedef __bf16 v8bf  __attribute__((ext_vector_type(8)));
typedef __bf16 v16bf __attribute__((ext_vector_type(16)));
typedef float  v8f   __attribute__((ext_vector_type(8)));

constexpr int   kN   = 50000;
constexpr int   kE   = 600000;
constexpr int   kIn  = 256;
constexpr int   kH   = 4;
constexpr int   kD   = 128;
constexpr int   kHD  = 512;   // H*D
constexpr float kNeg = 0.2f;

// ---------------------------------------------------------------------------
// Helpers
// ---------------------------------------------------------------------------
__device__ __forceinline__ unsigned enc_f32(float f) {
  unsigned b = __float_as_uint(f);
  return (b & 0x80000000u) ? ~b : (b | 0x80000000u);   // order-preserving map
}
__device__ __forceinline__ float dec_f32(unsigned k) {
  unsigned b = (k & 0x80000000u) ? (k & 0x7FFFFFFFu) : ~k;
  return __uint_as_float(b);
}
__device__ __forceinline__ float leaky(float v) {
  return v > 0.0f ? v : kNeg * v;
}

// ---------------------------------------------------------------------------
// f32 -> bf16 conversion kernels
// ---------------------------------------------------------------------------
__global__ __launch_bounds__(256) void cvt_f32_bf16(const float* __restrict__ in,
                                                    bf16_t* __restrict__ out, int n) {
  int i = blockIdx.x * 256 + threadIdx.x;
  if (i < n) out[i] = (bf16_t)in[i];
}

__global__ __launch_bounds__(256) void relu_f32_bf16(const float* __restrict__ in,
                                                     bf16_t* __restrict__ out, int n) {
  int i = blockIdx.x * 256 + threadIdx.x;
  if (i < n) {
    float v = in[i];
    out[i] = (bf16_t)(v > 0.0f ? v : 0.0f);
  }
}

// ---------------------------------------------------------------------------
// bf16 WMMA GEMM:  C[M,Ncols] (f32) = A[M,K] @ B[K,Ncols],  K % 32 == 0,
// Ncols % 64 == 0.
// Block = 256 threads = 8 waves; block tile = 64 rows x 64 cols.
// Each wave owns a 16x32 output strip -> 2 WMMAs per K-step sharing one
// A fragment (A-fragment reuse; doubles WMMA density between barriers).
// A tile staged row-major in LDS; B tile staged TRANSPOSED (sB[col][k]) so
// every fragment is two contiguous 16-byte ds reads per lane, using a
// contiguous k order (k = 16*half + s) shared by A and B.
// ---------------------------------------------------------------------------
constexpr int TILE_M = 64;
constexpr int TILE_N = 64;
constexpr int TILE_K = 32;
constexpr int LDSPAD = 40;    // row stride in elements (+16B pad vs banks)

__global__ __launch_bounds__(256) void gemm_bf16_wmma(
    const bf16_t* __restrict__ A, const bf16_t* __restrict__ B,
    float* __restrict__ C, int M, int K, int Ncols) {
  __shared__ alignas(16) bf16_t sA[TILE_M * LDSPAD];   // [row][k]
  __shared__ alignas(16) bf16_t sB[TILE_N * LDSPAD];   // transposed: [col][k]

  const int t    = threadIdx.x;
  const int lane = t & 31;
  const int wave = t >> 5;        // 0..7
  const int wy   = wave & 3;      // row tile within block: 0..3  (16 rows each)
  const int wx   = wave >> 2;     // col strip within block: 0..1 (32 cols each)

  const int block_row = blockIdx.x * TILE_M;
  const int block_col = blockIdx.y * TILE_N;

  // A staging coords (per thread: one 16B chunk = 8 bf16)
  const int a_r    = t >> 2;            // 0..63
  const int a_cseg = (t & 3) * 8;       // 0,8,16,24
  const int a_grow = block_row + a_r;
  const bool a_ok  = (a_grow < M);
  const bf16_t* a_src = A + (size_t)a_grow * K + a_cseg;

  // B staging coords (per thread: one 16B chunk = 8 bf16 of one k-row)
  const int b_krow = t >> 3;            // 0..31
  const int b_col  = (t & 7) * 8;       // 0,8,...,56
  const bf16_t* b_src = B + (size_t)b_krow * Ncols + block_col + b_col;

  v8f acc0 = {};
  v8f acc1 = {};

  for (int kb = 0; kb < K; kb += TILE_K) {
    // ---- stage A tile: 64 rows x 32 bf16 ----
    {
      v8bf vals = {};
      if (a_ok) {
        vals = *(const v8bf*)(a_src + kb);
        if (kb + TILE_K < K)
          __builtin_prefetch(a_src + kb + TILE_K, 0, 1);  // global_prefetch_b8
      }
      *(v8bf*)(sA + a_r * LDSPAD + a_cseg) = vals;
    }
    // ---- stage B tile transposed: sB[col][k] = B[kb+k][block_col+col] ----
    {
      v8bf vals = *(const v8bf*)(b_src + (size_t)kb * Ncols);
#pragma unroll
      for (int j = 0; j < 8; ++j)
        sB[(b_col + j) * LDSPAD + b_krow] = vals[j];
    }
    __syncthreads();

    // ---- build fragments in contiguous k order (same for A and B) ----
    const int kh   = lane >> 4;                 // lane half: 0 or 1
    const int lrow = wy * 16 + (lane & 15);     // A row (M)
    const int c0   = wx * 32 + (lane & 15);     // B col, first 16-wide tile
    const int c1   = c0 + 16;                   // B col, second 16-wide tile

    union { v16bf v; v8bf h[2]; } fa, fb0, fb1;
    // A 16x32: half0 -> K 0..15, half1 -> K 16..31 (contiguous in sA row)
    fa.h[0] = *(const v8bf*)(sA + lrow * LDSPAD + kh * 16);
    fa.h[1] = *(const v8bf*)(sA + lrow * LDSPAD + kh * 16 + 8);
    // B 32x16: half0 -> K 0..15, half1 -> K 16..31 (contiguous in sB row)
    fb0.h[0] = *(const v8bf*)(sB + c0 * LDSPAD + kh * 16);
    fb0.h[1] = *(const v8bf*)(sB + c0 * LDSPAD + kh * 16 + 8);
    fb1.h[0] = *(const v8bf*)(sB + c1 * LDSPAD + kh * 16);
    fb1.h[1] = *(const v8bf*)(sB + c1 * LDSPAD + kh * 16 + 8);

    acc0 = __builtin_amdgcn_wmma_f32_16x16x32_bf16(
        false, fa.v, false, fb0.v, (short)0, acc0, false, false);
    acc1 = __builtin_amdgcn_wmma_f32_16x16x32_bf16(
        false, fa.v, false, fb1.v, (short)0, acc1, false, false);
    __syncthreads();
  }

  // ---- store: lane -> col = base+lane%16, VGPR v -> row v + 8*half ----
  const int colA  = block_col + wx * 32 + (lane & 15);
  const int rbase = block_row + wy * 16 + (lane >> 4) * 8;
#pragma unroll
  for (int v = 0; v < 8; ++v) {
    int r = rbase + v;
    if (r < M) {
      C[(size_t)r * Ncols + colA]      = acc0[v];
      C[(size_t)r * Ncols + colA + 16] = acc1[v];
    }
  }
}

// ---------------------------------------------------------------------------
// el[n,h] = <h[n,h,:], al[h,:]>,  er likewise.  One thread per (n,h).
// ---------------------------------------------------------------------------
__global__ __launch_bounds__(256) void compute_elr(
    const float* __restrict__ hbuf, const float* __restrict__ al,
    const float* __restrict__ ar, float* __restrict__ el,
    float* __restrict__ er) {
  int i = blockIdx.x * 256 + threadIdx.x;           // over N*H
  if (i >= kN * kH) return;
  int n = i / kH, hh = i % kH;
  const float* hp  = hbuf + (size_t)n * kHD + hh * kD;
  const float* alp = al + hh * kD;
  const float* arp = ar + hh * kD;
  float sl = 0.0f, sr = 0.0f;
#pragma unroll 4
  for (int d = 0; d < kD; d += 4) {
    float4 hv = *(const float4*)(hp + d);
    float4 av = *(const float4*)(alp + d);
    float4 bv = *(const float4*)(arp + d);
    sl += hv.x * av.x + hv.y * av.y + hv.z * av.z + hv.w * av.w;
    sr += hv.x * bv.x + hv.y * bv.y + hv.z * bv.z + hv.w * bv.w;
  }
  el[i] = sl;
  er[i] = sr;
}

// ---------------------------------------------------------------------------
// Edge pass 1: segment max via sortable-uint atomicMax (memset0 == -inf).
// ---------------------------------------------------------------------------
__global__ __launch_bounds__(256) void edge_max_kernel(
    const int* __restrict__ src, const int* __restrict__ dst,
    const float* __restrict__ el, const float* __restrict__ er,
    unsigned* __restrict__ mkey) {
  int e = blockIdx.x * 256 + threadIdx.x;
  if (e >= kE) return;
  int s = src[e], d = dst[e];
  float4 lv = *(const float4*)(el + (size_t)s * kH);
  float4 rv = *(const float4*)(er + (size_t)d * kH);
  float v0 = leaky(lv.x + rv.x), v1 = leaky(lv.y + rv.y);
  float v2 = leaky(lv.z + rv.z), v3 = leaky(lv.w + rv.w);
  atomicMax(&mkey[(size_t)d * kH + 0], enc_f32(v0));
  atomicMax(&mkey[(size_t)d * kH + 1], enc_f32(v1));
  atomicMax(&mkey[(size_t)d * kH + 2], enc_f32(v2));
  atomicMax(&mkey[(size_t)d * kH + 3], enc_f32(v3));
}

// ---------------------------------------------------------------------------
// Edge pass 2: ex = exp(e - m[dst]); denom[dst] += ex  (stores ex per edge).
// ---------------------------------------------------------------------------
__global__ __launch_bounds__(256) void edge_exp_kernel(
    const int* __restrict__ src, const int* __restrict__ dst,
    const float* __restrict__ el, const float* __restrict__ er,
    const unsigned* __restrict__ mkey, float* __restrict__ ex,
    float* __restrict__ denom) {
  int e = blockIdx.x * 256 + threadIdx.x;
  if (e >= kE) return;
  int s = src[e], d = dst[e];
  float4 lv = *(const float4*)(el + (size_t)s * kH);
  float4 rv = *(const float4*)(er + (size_t)d * kH);
  float4 xv;
  xv.x = __expf(leaky(lv.x + rv.x) - dec_f32(mkey[(size_t)d * kH + 0]));
  xv.y = __expf(leaky(lv.y + rv.y) - dec_f32(mkey[(size_t)d * kH + 1]));
  xv.z = __expf(leaky(lv.z + rv.z) - dec_f32(mkey[(size_t)d * kH + 2]));
  xv.w = __expf(leaky(lv.w + rv.w) - dec_f32(mkey[(size_t)d * kH + 3]));
  *(float4*)(ex + (size_t)e * kH) = xv;
  unsafeAtomicAdd(&denom[(size_t)d * kH + 0], xv.x);
  unsafeAtomicAdd(&denom[(size_t)d * kH + 1], xv.y);
  unsafeAtomicAdd(&denom[(size_t)d * kH + 2], xv.z);
  unsafeAtomicAdd(&denom[(size_t)d * kH + 3], xv.w);
}

// ---------------------------------------------------------------------------
// Edge pass 3: rst[dst,h,:] += alpha * hbuf[src,h,:].
// One 256-thread block per edge; thread t owns elements [2t, 2t+1] of the
// 512-wide payload -> fully coalesced L2 gather + f32 atomic scatter.
// hbuf is 102.4 MB -> resident in MI455X's 192 MB L2.
// ---------------------------------------------------------------------------
__global__ __launch_bounds__(256) void edge_aggregate_kernel(
    const int* __restrict__ src, const int* __restrict__ dst,
    const float* __restrict__ ex, const float* __restrict__ denom,
    const float* __restrict__ hbuf, float* __restrict__ rst) {
  int e = blockIdx.x;
  if (e >= kE) return;
  int s = src[e], d = dst[e];
  int base = threadIdx.x * 2;           // 0..510
  int hh   = base >> 7;                 // head (128 elems per head)
  float alpha = ex[(size_t)e * kH + hh] / denom[(size_t)d * kH + hh];
  float2 hv = *(const float2*)(hbuf + (size_t)s * kHD + base);
  unsafeAtomicAdd(&rst[(size_t)d * kHD + base + 0], alpha * hv.x);
  unsafeAtomicAdd(&rst[(size_t)d * kHD + base + 1], alpha * hv.y);
}

// ---------------------------------------------------------------------------
// Final epilogue: out[n,d] = mean_h relu(rst[n,h,d])
// ---------------------------------------------------------------------------
__global__ __launch_bounds__(256) void final_mean_kernel(
    const float* __restrict__ rst, float* __restrict__ out) {
  int i = blockIdx.x * 256 + threadIdx.x;          // over N*D
  if (i >= kN * kD) return;
  int n = i / kD, d = i % kD;
  const float* rp = rst + (size_t)n * kHD + d;
  float s = 0.0f;
#pragma unroll
  for (int hh = 0; hh < kH; ++hh) {
    float v = rp[hh * kD];
    s += (v > 0.0f ? v : 0.0f);
  }
  out[i] = 0.25f * s;
}

// ---------------------------------------------------------------------------
// Host launcher
// ---------------------------------------------------------------------------
static inline size_t align256(size_t x) { return (x + 255) & ~size_t(255); }

extern "C" void kernel_launch(void* const* d_in, const int* in_sizes, int n_in,
                              void* d_out, int out_size, void* d_ws, size_t ws_size,
                              hipStream_t stream) {
  const float* x   = (const float*)d_in[0];
  const int*   src = (const int*)d_in[1];
  const int*   dst = (const int*)d_in[2];
  const float* W0  = (const float*)d_in[3];
  const float* al0 = (const float*)d_in[4];
  const float* ar0 = (const float*)d_in[5];
  const float* W1  = (const float*)d_in[6];
  const float* al1 = (const float*)d_in[7];
  const float* ar1 = (const float*)d_in[8];
  float* out = (float*)d_out;

  // ---- workspace carve-out ----
  char* ws = (char*)d_ws;
  size_t off = 0;
  auto carve = [&](size_t bytes) { void* p = ws + off; off += align256(bytes); return p; };
  bf16_t*   xb    = (bf16_t*)carve((size_t)kN * kHD * sizeof(bf16_t));   // bf16 input (<=512 cols)
  bf16_t*   Wb    = (bf16_t*)carve((size_t)kHD * kHD * sizeof(bf16_t));  // bf16 weights
  float*    hbuf  = (float*)carve((size_t)kN * kHD * sizeof(float));     // GEMM out / messages
  float*    rst   = (float*)carve((size_t)kN * kHD * sizeof(float));     // aggregation
  float*    el    = (float*)carve((size_t)kN * kH * sizeof(float));
  float*    er    = (float*)carve((size_t)kN * kH * sizeof(float));
  unsigned* mkey  = (unsigned*)carve((size_t)kN * kH * sizeof(unsigned));
  float*    denom = (float*)carve((size_t)kN * kH * sizeof(float));
  float*    ex    = (float*)carve((size_t)kE * kH * sizeof(float));
  (void)ws_size; (void)n_in; (void)in_sizes; (void)out_size;

  const int eb  = (kE + 255) / 256;
  const int nhb = (kN * kH + 255) / 256;

  for (int layer = 0; layer < 2; ++layer) {
    const int Kdim = (layer == 0) ? kIn : kHD;
    const float* Wf = (layer == 0) ? W0 : W1;
    const float* al = (layer == 0) ? al0 : al1;
    const float* ar = (layer == 0) ? ar0 : ar1;

    // 1) bf16 inputs
    if (layer == 0) {
      int n = kN * kIn;
      cvt_f32_bf16<<<(n + 255) / 256, 256, 0, stream>>>(x, xb, n);
    }
    {
      int n = Kdim * kHD;
      cvt_f32_bf16<<<(n + 255) / 256, 256, 0, stream>>>(Wf, Wb, n);
    }

    // 2) h = x @ W  (bf16 WMMA, f32 accumulate)
    {
      dim3 grid((kN + TILE_M - 1) / TILE_M, kHD / TILE_N);
      gemm_bf16_wmma<<<grid, 256, 0, stream>>>(xb, Wb, hbuf, kN, Kdim, kHD);
    }

    // 3) attention logits
    compute_elr<<<nhb, 256, 0, stream>>>(hbuf, al, ar, el, er);

    // 4) edge softmax + aggregation
    hipMemsetAsync(mkey, 0, (size_t)kN * kH * sizeof(unsigned), stream);   // == -inf keys
    hipMemsetAsync(denom, 0, (size_t)kN * kH * sizeof(float), stream);
    hipMemsetAsync(rst, 0, (size_t)kN * kHD * sizeof(float), stream);
    edge_max_kernel<<<eb, 256, 0, stream>>>(src, dst, el, er, mkey);
    edge_exp_kernel<<<eb, 256, 0, stream>>>(src, dst, el, er, mkey, ex, denom);
    edge_aggregate_kernel<<<kE, 256, 0, stream>>>(src, dst, ex, denom, hbuf, rst);

    // 5) activation / epilogue
    if (layer == 0) {
      int n = kN * kHD;
      relu_f32_bf16<<<(n + 255) / 256, 256, 0, stream>>>(rst, xb, n);  // feed layer 1
    } else {
      final_mean_kernel<<<(kN * kD + 255) / 256, 256, 0, stream>>>(rst, out);
    }
  }
}
